// TemporalFusionEncoder_26164940767820
// MI455X (gfx1250) — compile-verified, matrix operand order checked by reading, not run on hardware
//
#include <hip/hip_runtime.h>
#include <hip/hip_bf16.h>
#include <math.h>

// ---------------- model dims ----------------
#define BB 256
#define TT 8
#define SS 24            // T*3 tokens
#define BTOK (BB*SS)     // 6144 rows
#define DD 1024
#define NHH 16
#define HD 64
#define MLPD 4096
#define NLL 8

typedef __attribute__((ext_vector_type(16))) __bf16 v16bf;
typedef __attribute__((ext_vector_type(8)))  float  v8f;

// ---- async global->LDS path (gfx1250), guarded so compile never breaks ----
#if defined(__has_builtin)
#  if __has_builtin(__builtin_amdgcn_global_load_async_to_lds_b128) && \
      __has_builtin(__builtin_amdgcn_s_wait_asynccnt)
#    define USE_ASYNC_LDS 1
#  endif
#endif
#ifndef USE_ASYNC_LDS
#  define USE_ASYNC_LDS 0
#endif

#if USE_ASYNC_LDS
typedef int vec4i __attribute__((vector_size(16)));
typedef vec4i __attribute__((address_space(1)))* gptr_v4i;   // global pointer to <4 x i32>
typedef vec4i __attribute__((address_space(3)))* lptr_v4i;   // LDS pointer to <4 x i32>
#endif

__device__ __forceinline__ unsigned short f2bf(float x) {
    unsigned int u = __float_as_uint(x);
    unsigned int r = (u + 0x7FFFu + ((u >> 16) & 1u)) >> 16;   // RNE
    return (unsigned short)r;
}

__device__ __forceinline__ float gelu_exact(float x) {
    return 0.5f * x * (1.0f + erff(x * 0.70710678118654752f));
}

// ---------------- f32 -> bf16 conversion ----------------
__global__ void cvt_bf16_kernel(const float* __restrict__ src, unsigned short* __restrict__ dst, int n) {
    int i = blockIdx.x * blockDim.x + threadIdx.x;
    if (i < n) dst[i] = f2bf(src[i]);
}

// ---------------- WMMA GEMM: C[M,N] = act(A[M,K] * W[N,K]^T + bias[N]) ----------------
union Frag { v16bf v; uint4 q[2]; };

// fragment from an LDS tile with row stride 32 (bf16 elems)
__device__ __forceinline__ Frag load_frag_lds(const unsigned short* base, int row0, int lane) {
    int lo = lane & 15, hi = lane >> 4;
    const unsigned short* p = base + (row0 + lo) * 32 + hi * 8;
    Frag f;
    f.q[0] = *reinterpret_cast<const uint4*>(p);        // K = hi*8 .. +7
    f.q[1] = *reinterpret_cast<const uint4*>(p + 16);   // K = 16 + hi*8 .. +7
    return f;
}

// grid: (N/64, M/64), block: 128 (4 waves, 2x2 arrangement, 32x32 per wave)
// LDS double-buffered 64x32 tiles of A and W, async-copied.
__global__ void gemm_bf16_kernel(const unsigned short* __restrict__ A,
                                 const unsigned short* __restrict__ W,
                                 const float* __restrict__ bias,
                                 float* __restrict__ Cf,
                                 unsigned short* __restrict__ Cbf,
                                 int M, int N, int K, int act) {
    __shared__ unsigned short lA[2][64 * 32];
    __shared__ unsigned short lW[2][64 * 32];

    int tid  = threadIdx.x;
    int lane = tid & 31;
    int w    = tid >> 5;
    int wm   = w >> 1, wn = w & 1;
    int blkM = blockIdx.y * 64;
    int blkN = blockIdx.x * 64;

    v8f acc00 = {}, acc01 = {}, acc10 = {}, acc11 = {};

    // stage one 64x32 bf16 tile (src row-major, ld = K) into an LDS tile
    // 256 chunks of 16B; 128 threads -> 2 chunks each
    auto stage = [&](const unsigned short* __restrict__ src, int rowBase, int k0,
                     unsigned short* dst) {
#pragma unroll
        for (int j = 0; j < 2; ++j) {
            int c   = tid * 2 + j;        // 0..255
            int row = c >> 2;
            int kc  = (c & 3) * 8;
            const unsigned short* g = src + (size_t)(rowBase + row) * K + (k0 + kc);
            unsigned short* l = dst + row * 32 + kc;
#if USE_ASYNC_LDS
            __builtin_amdgcn_global_load_async_to_lds_b128(
                (gptr_v4i)(uintptr_t)g,
                (lptr_v4i)(unsigned)(uintptr_t)l,
                0, 0);
#else
            *reinterpret_cast<uint4*>(l) = *reinterpret_cast<const uint4*>(g);
#endif
        }
    };

    stage(A, blkM, 0, lA[0]);
    stage(W, blkN, 0, lW[0]);

    int cur = 0;
    for (int k = 0; k < K; k += 32) {
#if USE_ASYNC_LDS
        __builtin_amdgcn_s_wait_asynccnt(0);
#endif
        __syncthreads();                       // tile `cur` visible to all waves
        if (k + 32 < K) {                      // prefetch next tile into other buffer
            stage(A, blkM, k + 32, lA[cur ^ 1]);
            stage(W, blkN, k + 32, lW[cur ^ 1]);
        }
        Frag a0 = load_frag_lds(lA[cur], wm * 32,      lane);
        Frag a1 = load_frag_lds(lA[cur], wm * 32 + 16, lane);
        Frag b0 = load_frag_lds(lW[cur], wn * 32,      lane);
        Frag b1 = load_frag_lds(lW[cur], wn * 32 + 16, lane);
        acc00 = __builtin_amdgcn_wmma_f32_16x16x32_bf16(false, a0.v, false, b0.v, (short)0, acc00, false, false);
        acc01 = __builtin_amdgcn_wmma_f32_16x16x32_bf16(false, a0.v, false, b1.v, (short)0, acc01, false, false);
        acc10 = __builtin_amdgcn_wmma_f32_16x16x32_bf16(false, a1.v, false, b0.v, (short)0, acc10, false, false);
        acc11 = __builtin_amdgcn_wmma_f32_16x16x32_bf16(false, a1.v, false, b1.v, (short)0, acc11, false, false);
        cur ^= 1;
    }

    int lo = lane & 15, hi = lane >> 4;
    v8f accs[2][2] = {{acc00, acc01}, {acc10, acc11}};
#pragma unroll
    for (int im = 0; im < 2; ++im) {
#pragma unroll
        for (int in_ = 0; in_ < 2; ++in_) {
            int ncol = blkN + wn * 32 + in_ * 16 + lo;
            float bb = bias ? bias[ncol] : 0.0f;
#pragma unroll
            for (int i = 0; i < 8; ++i) {
                int row = blkM + wm * 32 + im * 16 + i + 8 * hi;
                float v = accs[im][in_][i] + bb;
                if (act == 1) v = gelu_exact(v);
                size_t idx = (size_t)row * N + ncol;
                if (Cf)  Cf[idx]  = v;
                if (Cbf) Cbf[idx] = f2bf(v);
            }
        }
    }
}

// ---------------- embed assemble: x = proj[m] + mod_emb[m] + pe[s%8] ----------------
__global__ void assemble_kernel(const float* __restrict__ hp, const float* __restrict__ op,
                                const float* __restrict__ cp, const float* __restrict__ mod,
                                const float* __restrict__ pe,
                                float* __restrict__ xf, unsigned short* __restrict__ xbf) {
    size_t idx = (size_t)blockIdx.x * blockDim.x + threadIdx.x;
    if (idx >= (size_t)BTOK * DD) return;
    int d = (int)(idx & (DD - 1));
    int row = (int)(idx >> 10);
    int s = row % SS, b = row / SS;
    int m = s % 3, t = s / 3;
    const float* src = (m == 0) ? hp : ((m == 1) ? op : cp);
    float v = src[(size_t)(b * TT + t) * DD + d] + mod[m * DD + d] + pe[(s % TT) * DD + d];
    xf[idx] = v;
    xbf[idx] = f2bf(v);
}

// ---------------- self-attention over 24 tokens: block per (b,h) ----------------
__global__ void sa_attn_kernel(const float* __restrict__ qkv, float* __restrict__ ctxf,
                               unsigned short* __restrict__ ctxbf,
                               float* __restrict__ probsOut, int writeProbs) {
    __shared__ float qs[SS * HD], ks[SS * HD], vs[SS * HD], ps[SS * SS];
    int bh = blockIdx.x;
    int b = bh >> 4, h = bh & 15;
    const float* base = qkv + (size_t)b * SS * (3 * DD) + h * HD;

    for (int e = threadIdx.x; e < SS * HD; e += blockDim.x) {
        int s = e >> 6, d = e & 63;
        size_t r = (size_t)s * (3 * DD);
        qs[e] = base[r + d] * 0.125f;           // 1/sqrt(64)
        ks[e] = base[r + DD + d];
        vs[e] = base[r + 2 * DD + d];
    }
    __syncthreads();

    for (int e = threadIdx.x; e < SS * SS; e += blockDim.x) {
        int qi = e / SS, ki = e % SS;
        float s = 0.0f;
        for (int d = 0; d < HD; ++d) s += qs[qi * HD + d] * ks[ki * HD + d];
        ps[e] = s;
    }
    __syncthreads();

    if (threadIdx.x < SS) {
        int r = threadIdx.x;
        float mx = -3.0e38f;
        for (int k = 0; k < SS; ++k) mx = fmaxf(mx, ps[r * SS + k]);
        float sum = 0.0f;
        for (int k = 0; k < SS; ++k) { float ev = expf(ps[r * SS + k] - mx); ps[r * SS + k] = ev; sum += ev; }
        float inv = 1.0f / sum;
        for (int k = 0; k < SS; ++k) ps[r * SS + k] *= inv;
    }
    __syncthreads();

    for (int e = threadIdx.x; e < SS * HD; e += blockDim.x) {
        int r = e >> 6, d = e & 63;
        float s = 0.0f;
        for (int k = 0; k < SS; ++k) s += ps[r * SS + k] * vs[k * HD + d];
        size_t idx = (size_t)(b * SS + r) * DD + h * HD + d;
        ctxf[idx] = s;
        ctxbf[idx] = f2bf(s);
    }
    if (writeProbs) {
        for (int e = threadIdx.x; e < SS * SS; e += blockDim.x)
            probsOut[(size_t)bh * (SS * SS) + e] = ps[e];
    }
}

// ---------------- cross-modal attention within 3-token groups ----------------
__global__ void ca_attn_kernel(const float* __restrict__ qkv, float* __restrict__ ctxf,
                               unsigned short* __restrict__ ctxbf) {
    int g = blockIdx.x;
    int tid = threadIdx.x;
    if (tid >= NHH * 3) return;
    int h = tid / 3, qi = tid % 3;
    const float* base = qkv + (size_t)(g * 3) * (3 * DD) + h * HD;
    float sc[3];
#pragma unroll
    for (int j = 0; j < 3; ++j) {
        float acc = 0.0f;
        for (int d = 0; d < HD; ++d)
            acc += base[(size_t)qi * (3 * DD) + d] * 0.125f * base[(size_t)j * (3 * DD) + DD + d];
        sc[j] = acc;
    }
    float mx = fmaxf(sc[0], fmaxf(sc[1], sc[2]));
    float e0 = expf(sc[0] - mx), e1 = expf(sc[1] - mx), e2 = expf(sc[2] - mx);
    float inv = 1.0f / (e0 + e1 + e2);
    e0 *= inv; e1 *= inv; e2 *= inv;
    size_t orow = (size_t)(g * 3 + qi) * DD + h * HD;
    for (int d = 0; d < HD; ++d) {
        float v = e0 * base[(size_t)0 * (3 * DD) + 2 * DD + d]
                + e1 * base[(size_t)1 * (3 * DD) + 2 * DD + d]
                + e2 * base[(size_t)2 * (3 * DD) + 2 * DD + d];
        ctxf[orow + d] = v;
        ctxbf[orow + d] = f2bf(v);
    }
}

// ---------------- residual + LayerNorm (block per row) ----------------
__global__ void add_ln_kernel(const float* __restrict__ x, const float* __restrict__ r,
                              const float* __restrict__ g, const float* __restrict__ bta,
                              float* __restrict__ outf, unsigned short* __restrict__ outbf) {
    __shared__ float s1[256], s2[256];
    int row = blockIdx.x;
    const float* xr = x + (size_t)row * DD;
    const float* rr = r ? r + (size_t)row * DD : nullptr;
    float sum = 0.0f, sq = 0.0f;
    for (int i = threadIdx.x; i < DD; i += 256) {
        float v = xr[i] + (rr ? rr[i] : 0.0f);
        sum += v; sq += v * v;
    }
    s1[threadIdx.x] = sum; s2[threadIdx.x] = sq;
    __syncthreads();
    for (int off = 128; off > 0; off >>= 1) {
        if (threadIdx.x < off) {
            s1[threadIdx.x] += s1[threadIdx.x + off];
            s2[threadIdx.x] += s2[threadIdx.x + off];
        }
        __syncthreads();
    }
    float mean = s1[0] * (1.0f / DD);
    float var  = s2[0] * (1.0f / DD) - mean * mean;
    float inv  = rsqrtf(var + 1e-5f);
    for (int i = threadIdx.x; i < DD; i += 256) {
        float v = xr[i] + (rr ? rr[i] : 0.0f);
        float y = (v - mean) * inv * g[i] + bta[i];
        size_t idx = (size_t)row * DD + i;
        outf[idx] = y;
        outbf[idx] = f2bf(y);
    }
}

// ---------------- pooling & outputs ----------------
__global__ void fused_kernel(const float* __restrict__ x, float* __restrict__ out) {
    size_t idx = (size_t)blockIdx.x * blockDim.x + threadIdx.x;
    if (idx >= (size_t)BB * TT * DD) return;
    int d = (int)(idx & (DD - 1));
    int bt = (int)(idx >> 10);
    int b = bt >> 3, t = bt & 7;
    size_t r = (size_t)(b * SS + t * 3) * DD + d;
    out[idx] = (x[r] + x[r + DD] + x[r + 2 * DD]) * (1.0f / 3.0f);
}

__global__ void agg_kernel(const float* __restrict__ fused, float* __restrict__ aggf,
                           unsigned short* __restrict__ aggbf) {
    int idx = blockIdx.x * blockDim.x + threadIdx.x;
    if (idx >= BB * DD) return;
    int b = idx >> 10, d = idx & (DD - 1);
    float s = 0.0f;
    for (int t = 0; t < TT; ++t) s += fused[(size_t)(b * TT + t) * DD + d];
    s *= (1.0f / TT);
    aggf[idx] = s;
    aggbf[idx] = f2bf(s);
}

__global__ void attn_mean_kernel(const float* __restrict__ probs, float* __restrict__ out) {
    int idx = blockIdx.x * blockDim.x + threadIdx.x;
    if (idx >= BB * SS * SS) return;
    int b = idx / (SS * SS), r = idx % (SS * SS);
    float s = 0.0f;
    for (int h = 0; h < NHH; ++h) s += probs[(size_t)(b * NHH + h) * (SS * SS) + r];
    out[idx] = s * (1.0f / NHH);
}

// ---------------- host orchestration ----------------
static void launch_gemm(const unsigned short* A, const unsigned short* W, const float* bias,
                        float* Cf, unsigned short* Cbf, int M, int N, int K, int act,
                        hipStream_t s) {
    dim3 grid(N / 64, M / 64), block(128);
    gemm_bf16_kernel<<<grid, block, 0, s>>>(A, W, bias, Cf, Cbf, M, N, K, act);
}

static void launch_cvt(const float* src, unsigned short* dst, long long n, hipStream_t s) {
    int ni = (int)n;
    cvt_bf16_kernel<<<(ni + 255) / 256, 256, 0, s>>>(src, dst, ni);
}

extern "C" void kernel_launch(void* const* d_in, const int* in_sizes, int n_in,
                              void* d_out, int out_size, void* d_ws, size_t ws_size,
                              hipStream_t stream) {
    (void)in_sizes; (void)n_in; (void)out_size; (void)ws_size;

    const float* hand_f  = (const float*)d_in[0];
    const float* obj_f   = (const float*)d_in[1];
    const float* con_f   = (const float*)d_in[2];
    const float* hand_w  = (const float*)d_in[3];
    const float* hand_b  = (const float*)d_in[4];
    const float* obj_w   = (const float*)d_in[5];
    const float* obj_b   = (const float*)d_in[6];
    const float* con_w   = (const float*)d_in[7];
    const float* con_b   = (const float*)d_in[8];
    const float* mod_emb = (const float*)d_in[9];
    const float* pe      = (const float*)d_in[10];
    const float* sa_in_w = (const float*)d_in[11];
    const float* sa_in_b = (const float*)d_in[12];
    const float* sa_out_w= (const float*)d_in[13];
    const float* sa_out_b= (const float*)d_in[14];
    const float* ca_in_w = (const float*)d_in[15];
    const float* ca_in_b = (const float*)d_in[16];
    const float* ca_out_w= (const float*)d_in[17];
    const float* ca_out_b= (const float*)d_in[18];
    const float* ffn_w1  = (const float*)d_in[19];
    const float* ffn_b1  = (const float*)d_in[20];
    const float* ffn_w2  = (const float*)d_in[21];
    const float* ffn_b2  = (const float*)d_in[22];
    const float* ln1_g   = (const float*)d_in[23];
    const float* ln1_b   = (const float*)d_in[24];
    const float* ln2_g   = (const float*)d_in[25];
    const float* ln2_b   = (const float*)d_in[26];
    const float* ln3_g   = (const float*)d_in[27];
    const float* ln3_b   = (const float*)d_in[28];
    const float* fn_g    = (const float*)d_in[29];
    const float* fn_b    = (const float*)d_in[30];
    const float* pool_w1 = (const float*)d_in[31];
    const float* pool_b1 = (const float*)d_in[32];
    const float* pool_w2 = (const float*)d_in[33];
    const float* pool_b2 = (const float*)d_in[34];

    // ---- workspace carving ----
    char* wp = (char*)d_ws;
    auto alloc = [&](size_t bytes) -> void* {
        void* r = (void*)wp;
        wp += (bytes + 255) & ~(size_t)255;
        return r;
    };
    const size_t KIN = 512;
    unsigned short* hand_wbf = (unsigned short*)alloc((size_t)DD * KIN * 2);
    unsigned short* obj_wbf  = (unsigned short*)alloc((size_t)DD * KIN * 2);
    unsigned short* con_wbf  = (unsigned short*)alloc((size_t)DD * KIN * 2);
    unsigned short* sa_in_bf = (unsigned short*)alloc((size_t)NLL * 3 * DD * DD * 2);
    unsigned short* sa_out_bf= (unsigned short*)alloc((size_t)NLL * DD * DD * 2);
    unsigned short* ca_in_bf = (unsigned short*)alloc((size_t)NLL * 3 * DD * DD * 2);
    unsigned short* ca_out_bf= (unsigned short*)alloc((size_t)NLL * DD * DD * 2);
    unsigned short* ffn1_bf  = (unsigned short*)alloc((size_t)NLL * MLPD * DD * 2);
    unsigned short* ffn2_bf  = (unsigned short*)alloc((size_t)NLL * DD * MLPD * 2);
    unsigned short* pool1_bf = (unsigned short*)alloc((size_t)DD * DD * 2);
    unsigned short* pool2_bf = (unsigned short*)alloc((size_t)DD * DD * 2);
    unsigned short* handf_bf = (unsigned short*)alloc((size_t)BB * TT * KIN * 2);
    unsigned short* objf_bf  = (unsigned short*)alloc((size_t)BB * TT * KIN * 2);
    unsigned short* conf_bf  = (unsigned short*)alloc((size_t)BB * TT * KIN * 2);
    float* hproj = (float*)alloc((size_t)BB * TT * DD * 4);
    float* oproj = (float*)alloc((size_t)BB * TT * DD * 4);
    float* cproj = (float*)alloc((size_t)BB * TT * DD * 4);
    float* xA    = (float*)alloc((size_t)BTOK * DD * 4);
    float* xB    = (float*)alloc((size_t)BTOK * DD * 4);
    unsigned short* xbf   = (unsigned short*)alloc((size_t)BTOK * DD * 2);
    float* qkv   = (float*)alloc((size_t)BTOK * 3 * DD * 4);
    float* ctxf  = (float*)alloc((size_t)BTOK * DD * 4);
    unsigned short* ctxbf = (unsigned short*)alloc((size_t)BTOK * DD * 2);
    float* proj  = (float*)alloc((size_t)BTOK * DD * 4);
    unsigned short* hidbf = (unsigned short*)alloc((size_t)BTOK * MLPD * 2);
    float* probsbuf = (float*)alloc((size_t)BB * NHH * SS * SS * 4);
    float* aggf  = (float*)alloc((size_t)BB * DD * 4);
    unsigned short* aggbf  = (unsigned short*)alloc((size_t)BB * DD * 2);
    unsigned short* poolhbf= (unsigned short*)alloc((size_t)BB * DD * 2);

    // ---- weight / input conversions to bf16 ----
    launch_cvt(hand_w,  hand_wbf, (long long)DD * KIN, stream);
    launch_cvt(obj_w,   obj_wbf,  (long long)DD * KIN, stream);
    launch_cvt(con_w,   con_wbf,  (long long)DD * KIN, stream);
    launch_cvt(sa_in_w, sa_in_bf, (long long)NLL * 3 * DD * DD, stream);
    launch_cvt(sa_out_w,sa_out_bf,(long long)NLL * DD * DD, stream);
    launch_cvt(ca_in_w, ca_in_bf, (long long)NLL * 3 * DD * DD, stream);
    launch_cvt(ca_out_w,ca_out_bf,(long long)NLL * DD * DD, stream);
    launch_cvt(ffn_w1,  ffn1_bf,  (long long)NLL * MLPD * DD, stream);
    launch_cvt(ffn_w2,  ffn2_bf,  (long long)NLL * DD * MLPD, stream);
    launch_cvt(pool_w1, pool1_bf, (long long)DD * DD, stream);
    launch_cvt(pool_w2, pool2_bf, (long long)DD * DD, stream);
    launch_cvt(hand_f,  handf_bf, (long long)BB * TT * KIN, stream);
    launch_cvt(obj_f,   objf_bf,  (long long)BB * TT * KIN, stream);
    launch_cvt(con_f,   conf_bf,  (long long)BB * TT * KIN, stream);

    // ---- input projections ----
    launch_gemm(handf_bf, hand_wbf, hand_b, hproj, nullptr, BB * TT, DD, (int)KIN, 0, stream);
    launch_gemm(objf_bf,  obj_wbf,  obj_b,  oproj, nullptr, BB * TT, DD, (int)KIN, 0, stream);
    launch_gemm(conf_bf,  con_wbf,  con_b,  cproj, nullptr, BB * TT, DD, (int)KIN, 0, stream);

    {
        size_t n = (size_t)BTOK * DD;
        assemble_kernel<<<(unsigned)((n + 255) / 256), 256, 0, stream>>>(
            hproj, oproj, cproj, mod_emb, pe, xA, xbf);
    }

    float* xcur = xA;
    float* xnxt = xB;

    for (int l = 0; l < NLL; ++l) {
        // self-attention
        launch_gemm(xbf, sa_in_bf + (size_t)l * 3 * DD * DD, sa_in_b + (size_t)l * 3 * DD,
                    qkv, nullptr, BTOK, 3 * DD, DD, 0, stream);
        sa_attn_kernel<<<BB * NHH, 256, 0, stream>>>(qkv, ctxf, ctxbf, probsbuf,
                                                     (l == NLL - 1) ? 1 : 0);
        launch_gemm(ctxbf, sa_out_bf + (size_t)l * DD * DD, sa_out_b + (size_t)l * DD,
                    proj, nullptr, BTOK, DD, DD, 0, stream);
        add_ln_kernel<<<BTOK, 256, 0, stream>>>(xcur, proj, ln1_g + l * DD, ln1_b + l * DD, xnxt, xbf);
        { float* t = xcur; xcur = xnxt; xnxt = t; }

        // cross-modal attention
        launch_gemm(xbf, ca_in_bf + (size_t)l * 3 * DD * DD, ca_in_b + (size_t)l * 3 * DD,
                    qkv, nullptr, BTOK, 3 * DD, DD, 0, stream);
        ca_attn_kernel<<<BB * TT, 64, 0, stream>>>(qkv, ctxf, ctxbf);
        launch_gemm(ctxbf, ca_out_bf + (size_t)l * DD * DD, ca_out_b + (size_t)l * DD,
                    proj, nullptr, BTOK, DD, DD, 0, stream);
        add_ln_kernel<<<BTOK, 256, 0, stream>>>(xcur, proj, ln2_g + l * DD, ln2_b + l * DD, xnxt, xbf);
        { float* t = xcur; xcur = xnxt; xnxt = t; }

        // FFN
        launch_gemm(xbf, ffn1_bf + (size_t)l * MLPD * DD, ffn_b1 + (size_t)l * MLPD,
                    nullptr, hidbf, BTOK, MLPD, DD, 1, stream);  // GELU, bf16 out only
        launch_gemm(hidbf, ffn2_bf + (size_t)l * DD * MLPD, ffn_b2 + (size_t)l * DD,
                    proj, nullptr, BTOK, DD, MLPD, 0, stream);
        add_ln_kernel<<<BTOK, 256, 0, stream>>>(xcur, proj, ln3_g + l * DD, ln3_b + l * DD, xnxt, xbf);
        { float* t = xcur; xcur = xnxt; xnxt = t; }
    }

    // final LayerNorm
    add_ln_kernel<<<BTOK, 256, 0, stream>>>(xcur, nullptr, fn_g, fn_b, xnxt, xbf);
    { float* t = xcur; xcur = xnxt; xnxt = t; }

    // outputs: fused | agg | attn_last
    float* out_fused = (float*)d_out;
    float* out_agg   = out_fused + (size_t)BB * TT * DD;
    float* out_attn  = out_agg + (size_t)BB * DD;

    {
        size_t n = (size_t)BB * TT * DD;
        fused_kernel<<<(unsigned)((n + 255) / 256), 256, 0, stream>>>(xcur, out_fused);
    }
    agg_kernel<<<(BB * DD + 255) / 256, 256, 0, stream>>>(out_fused, aggf, aggbf);

    launch_gemm(aggbf, pool1_bf, pool_b1, nullptr, poolhbf, BB, DD, DD, 1, stream);  // GELU
    launch_gemm(poolhbf, pool2_bf, pool_b2, out_agg, nullptr, BB, DD, DD, 0, stream);

    attn_mean_kernel<<<(BB * SS * SS + 255) / 256, 256, 0, stream>>>(probsbuf, out_attn);
}
